// AdaptIR_50732153700575
// MI455X (gfx1250) — compile-verified, matrix operand order checked by reading, not run on hardware
//
#include <hip/hip_runtime.h>
#include <math.h>

// ---------------------------------------------------------------------------
// AdaptIR forward for MI455X (gfx1250, wave32, WMMA)
//   N=8, H=W=64, D_MODEL=896, HIDDEN=64, RANK=32
// ---------------------------------------------------------------------------

typedef __attribute__((ext_vector_type(16))) _Float16 v16h;
typedef __attribute__((ext_vector_type(8)))  float    v8f;

#define NBATCH   8
#define HID      64
#define DMODEL   896
#define HW       4096          // 64*64
#define MROWS    32768         // NBATCH*HW

__device__ __forceinline__ float gelu_f(float x) {
    return 0.5f * x * (1.0f + erff(x * 0.70710678118654752f));
}
__device__ __forceinline__ float sigmoid_f(float x) {
    return 1.0f / (1.0f + expf(-x));
}

// Load a 16x32 fragment (A or B^T) from a row-major f32 matrix, converting to
// f16.  Per CDNA5 ISA 7.12.2 (16-bit A 16x32): lane<16 -> row=lane, K={0..7}
// in v0..3 and K={16..23} in v4..7; lane>=16 -> row=lane-16, K={8..15,24..31}.
__device__ __forceinline__ v16h load_frag_f32(const float* __restrict__ base,
                                              int lda, int lane) {
    const int r  = lane & 15;
    const int kb = (lane < 16) ? 0 : 8;
    const float* p = base + (size_t)r * lda + kb;
    v16h f;
#pragma unroll
    for (int i = 0; i < 8; ++i) f[i]     = (_Float16)p[i];
#pragma unroll
    for (int i = 0; i < 8; ++i) f[8 + i] = (_Float16)p[16 + i];
    return f;
}

// ---------------------------------------------------------------------------
// K0: tiny prep — bn_scale, low-rank depthwise weights, DFT twiddle table
// ---------------------------------------------------------------------------
__global__ void __launch_bounds__(64)
prep_kernel(const float* __restrict__ bn_w, const float* __restrict__ bn_var,
            const float* __restrict__ conv_A, const float* __restrict__ conv_B,
            float* __restrict__ bn_scale, float* __restrict__ w9,
            float* __restrict__ twc, float* __restrict__ tws) {
    const int t = threadIdx.x;          // 64 threads
    bn_scale[t] = bn_w[t] * rsqrtf(bn_var[t] + 1e-5f);
#pragma unroll
    for (int j = 0; j < 9; ++j) {
        float acc = 0.f;
        for (int r = 0; r < 32; ++r) acc += conv_A[t * 32 + r] * conv_B[r * 9 + j];
        w9[t * 9 + j] = acc;
    }
    const float th = 6.28318530717958647f * (float)t * (1.0f / 64.0f);
    twc[t] = cosf(th);
    tws[t] = sinf(th);
}

// ---------------------------------------------------------------------------
// K1: head GEMM  xc = BN( x[32768,896] @ head_w^T + head_b )  -> NCHW planes
//     One wave owns one 16-row m-tile and ALL 4 n-tiles (x read exactly once
//     from HBM; head_w streams from L2).  WMMA f32 <- f16, K steps of 32.
// ---------------------------------------------------------------------------
__global__ void __launch_bounds__(256)
head_gemm_kernel(const float* __restrict__ X, const float* __restrict__ Wt,
                 const float* __restrict__ head_b,
                 const float* __restrict__ bn_scale,
                 const float* __restrict__ bn_mean,
                 const float* __restrict__ bn_b,
                 float* __restrict__ xc) {
    const int lane = threadIdx.x & 31;
    const int wave = threadIdx.x >> 5;
    const int mt = blockIdx.x * 8 + wave;       // 2048 m-tiles
    const int m0 = mt << 4;

    v8f acc0 = {0.f,0.f,0.f,0.f,0.f,0.f,0.f,0.f};
    v8f acc1 = acc0, acc2 = acc0, acc3 = acc0;

    const float* abase = X + (size_t)m0 * DMODEL;
#pragma unroll 2
    for (int k0 = 0; k0 < DMODEL; k0 += 32) {
        __builtin_prefetch(abase + k0 + 32, 0, 1);   // global_prefetch_b8
        v16h a  = load_frag_f32(abase + k0, DMODEL, lane);
        v16h b0 = load_frag_f32(Wt + k0,                         DMODEL, lane);
        v16h b1 = load_frag_f32(Wt + (size_t)16 * DMODEL + k0,   DMODEL, lane);
        v16h b2 = load_frag_f32(Wt + (size_t)32 * DMODEL + k0,   DMODEL, lane);
        v16h b3 = load_frag_f32(Wt + (size_t)48 * DMODEL + k0,   DMODEL, lane);
        acc0 = __builtin_amdgcn_wmma_f32_16x16x32_f16(false, a, false, b0,
                                                      (short)0, acc0, false, false);
        acc1 = __builtin_amdgcn_wmma_f32_16x16x32_f16(false, a, false, b1,
                                                      (short)0, acc1, false, false);
        acc2 = __builtin_amdgcn_wmma_f32_16x16x32_f16(false, a, false, b2,
                                                      (short)0, acc2, false, false);
        acc3 = __builtin_amdgcn_wmma_f32_16x16x32_f16(false, a, false, b3,
                                                      (short)0, acc3, false, false);
    }

    // C layout: VGPR r -> M = m0 + r (+8 for lanes 16..31), N = lane&15
    const int mrow = m0 + ((lane < 16) ? 0 : 8);
    const int n  = mrow >> 12;
    const int hw = mrow & 4095;
    v8f accs[4] = {acc0, acc1, acc2, acc3};
#pragma unroll
    for (int nt = 0; nt < 4; ++nt) {
        const int o = (nt << 4) + (lane & 15);
        const float hb = head_b[o], sc = bn_scale[o];
        const float mn = bn_mean[o], bb = bn_b[o];
        float* dst = xc + ((size_t)(n * HID + o) << 12) + hw;
#pragma unroll
        for (int r = 0; r < 8; ++r)
            dst[r] = (accs[nt][r] + hb - mn) * sc + bb;
    }
}

// ---------------------------------------------------------------------------
// K2: frequency branch — per (n,c) plane, fully LDS-resident direct DFTs.
//     rfft2(ortho) -> amp/phase affine -> hermitian extend -> ifft2 -> |.|
//     Also produces the per-plane spatial mean (no atomics needed).
// ---------------------------------------------------------------------------
__global__ void __launch_bounds__(256)
fft_branch_kernel(const float* __restrict__ xc,
                  const float* __restrict__ amp_w, const float* __restrict__ amp_b,
                  const float* __restrict__ pha_w, const float* __restrict__ pha_b,
                  const float* __restrict__ twc, const float* __restrict__ tws,
                  float* __restrict__ global_x, float* __restrict__ avg) {
    __shared__ float Ar[4096], Ai[4096], Br[4096], Bi[4096];   // 64 KB
    const int tid = threadIdx.x;
    const int plane = blockIdx.x;                 // n*64 + c
    const int c = plane & 63;
    const float aw = amp_w[c], ab = amp_b[c], pw = pha_w[c], pb = pha_b[c];
    const float* src = xc + ((size_t)plane << 12);

    for (int i = tid; i < 4096; i += 256) Ar[i] = src[i];
    __syncthreads();

    // P1: forward DFT along rows (w -> kx), real input
    for (int i = tid; i < 4096; i += 256) {
        const int r = i >> 6, k = i & 63;
        const float* row = &Ar[r << 6];
        float re = 0.f, im = 0.f;
        for (int w = 0; w < 64; ++w) {
            const int j = (k * w) & 63;
            const float v = row[w];
            re += v * twc[j];
            im -= v * tws[j];
        }
        Br[i] = re; Bi[i] = im;
    }
    __syncthreads();
    // P2: forward DFT along cols (r -> ky), ortho scale 1/64
    for (int i = tid; i < 4096; i += 256) {
        const int ky = i >> 6, kx = i & 63;
        float re = 0.f, im = 0.f;
        for (int r = 0; r < 64; ++r) {
            const int j = (ky * r) & 63;
            const float br = Br[(r << 6) + kx], bi = Bi[(r << 6) + kx];
            const float cc = twc[j], ss = tws[j];
            re += br * cc + bi * ss;     // * e^{-i th}
            im += bi * cc - br * ss;
        }
        Ar[i] = re * 0.015625f; Ai[i] = im * 0.015625f;
    }
    __syncthreads();
    // P3a: amp/phase affine on kept half (kx <= 32)
    for (int i = tid; i < 4096; i += 256) {
        const int kx = i & 63;
        if (kx <= 32) {
            const float re = Ar[i], im = Ai[i];
            const float mag = sqrtf(re * re + im * im) * aw + ab;
            const float ph  = atan2f(im, re) * pw + pb;
            Br[i] = mag * cosf(ph);
            Bi[i] = mag * sinf(ph);
        }
    }
    __syncthreads();
    // P3b: hermitian mirror: G[ky][kx] = conj(G[(64-ky)%64][64-kx]) for kx>=33
    for (int i = tid; i < 4096; i += 256) {
        const int ky = i >> 6, kx = i & 63;
        if (kx >= 33) {
            const int j = (((64 - ky) & 63) << 6) + (64 - kx);
            Br[i] =  Br[j];
            Bi[i] = -Bi[j];
        }
    }
    __syncthreads();
    // P4: inverse DFT along rows (kx -> w)
    for (int i = tid; i < 4096; i += 256) {
        const int ky = i >> 6, w = i & 63;
        const int rb = ky << 6;
        float re = 0.f, im = 0.f;
        for (int kx = 0; kx < 64; ++kx) {
            const int j = (kx * w) & 63;
            const float br = Br[rb + kx], bi = Bi[rb + kx];
            const float cc = twc[j], ss = tws[j];
            re += br * cc - bi * ss;     // * e^{+i th}
            im += br * ss + bi * cc;
        }
        Ar[i] = re; Ai[i] = im;
    }
    __syncthreads();
    // P5: inverse DFT along cols, real part, ortho 1/64, abs, mean-accumulate
    float psum = 0.f;
    float* dst = global_x + ((size_t)plane << 12);
    for (int i = tid; i < 4096; i += 256) {
        const int r = i >> 6, w = i & 63;
        float re = 0.f;
        for (int ky = 0; ky < 64; ++ky) {
            const int j = (ky * r) & 63;
            re += Ar[(ky << 6) + w] * twc[j] - Ai[(ky << 6) + w] * tws[j];
        }
        const float v = fabsf(re * 0.015625f);
        dst[i] = v;
        psum += v;
    }
    __syncthreads();
    Br[tid] = psum;
    __syncthreads();
    for (int s = 128; s > 0; s >>= 1) {
        if (tid < s) Br[tid] += Br[tid + s];
        __syncthreads();
    }
    if (tid == 0) avg[plane] = Br[0] * (1.0f / 4096.0f);
}

// ---------------------------------------------------------------------------
// K3: low-rank depthwise 3x3 conv (weights precomputed by prep)
// ---------------------------------------------------------------------------
__global__ void __launch_bounds__(256)
dwconv_kernel(const float* __restrict__ xc, const float* __restrict__ w9,
              const float* __restrict__ conv_bias, float* __restrict__ local_x) {
    const int idx = blockIdx.x * 256 + threadIdx.x;    // 2,097,152
    const int plane = idx >> 12;
    const int c = plane & 63;
    const int y = (idx >> 6) & 63;
    const int x = idx & 63;
    const float* p = xc + ((size_t)plane << 12);
    const float* w = w9 + c * 9;
    float acc = conv_bias[c];
#pragma unroll
    for (int dy = -1; dy <= 1; ++dy) {
        const int yy = y + dy;
        if (yy < 0 || yy > 63) continue;
#pragma unroll
        for (int dx = -1; dx <= 1; ++dx) {
            const int xx = x + dx;
            if (xx < 0 || xx > 63) continue;
            acc += w[(dy + 1) * 3 + (dx + 1)] * p[(yy << 6) + xx];
        }
    }
    local_x[idx] = acc;
}

// ---------------------------------------------------------------------------
// K4: attention pooling: score = softmax_HW(compress(xc)); pooled = xc . score
// ---------------------------------------------------------------------------
__global__ void __launch_bounds__(256)
pool_kernel(const float* __restrict__ xc, const float* __restrict__ compress_w,
            const float* __restrict__ compress_b, float* __restrict__ pooled) {
    __shared__ float s[4096];
    __shared__ float red[256];
    const int n = blockIdx.x, tid = threadIdx.x;
    const float cb = compress_b[0];
    const float* base = xc + ((size_t)n * HID << 12);
    for (int i = tid; i < 4096; i += 256) {
        float acc = cb;
        for (int c = 0; c < HID; ++c) acc += compress_w[c] * base[((size_t)c << 12) + i];
        s[i] = acc;
    }
    __syncthreads();
    float m = -INFINITY;
    for (int i = tid; i < 4096; i += 256) m = fmaxf(m, s[i]);
    red[tid] = m; __syncthreads();
    for (int st = 128; st > 0; st >>= 1) {
        if (tid < st) red[tid] = fmaxf(red[tid], red[tid + st]);
        __syncthreads();
    }
    m = red[0]; __syncthreads();
    float sum = 0.f;
    for (int i = tid; i < 4096; i += 256) { const float e = expf(s[i] - m); s[i] = e; sum += e; }
    red[tid] = sum; __syncthreads();
    for (int st = 128; st > 0; st >>= 1) {
        if (tid < st) red[tid] += red[tid + st];
        __syncthreads();
    }
    const float inv = 1.0f / red[0];
    __syncthreads();
    for (int c = 0; c < HID; ++c) {
        float acc = 0.f;
        const float* p = base + ((size_t)c << 12);
        for (int i = tid; i < 4096; i += 256) acc += p[i] * s[i];
        red[tid] = acc; __syncthreads();
        for (int st = 128; st > 0; st >>= 1) {
            if (tid < st) red[tid] += red[tid + st];
            __syncthreads();
        }
        if (tid == 0) pooled[n * HID + c] = red[0] * inv;
        __syncthreads();
    }
}

// ---------------------------------------------------------------------------
// K5: spatial gate = sigmoid(si . local_x)
// ---------------------------------------------------------------------------
__global__ void __launch_bounds__(256)
sgate_kernel(const float* __restrict__ local_x, const float* __restrict__ si_w,
             const float* __restrict__ si_b, float* __restrict__ sg) {
    const int idx = blockIdx.x * 256 + threadIdx.x;   // 32768
    const int n = idx >> 12, hw = idx & 4095;
    const float* p = local_x + ((size_t)n << 18) + hw;
    float acc = si_b[0];
    for (int c = 0; c < HID; ++c) acc += si_w[c] * p[(size_t)c << 12];
    sg[idx] = sigmoid_f(acc);
}

// ---------------------------------------------------------------------------
// K6: tiny MLPs -> channel_score, channel_gate   (one block per n, 64 thr)
// ---------------------------------------------------------------------------
__global__ void __launch_bounds__(64)
mlp_kernel(const float* __restrict__ pooled, const float* __restrict__ avg,
           const float* __restrict__ proj_w1, const float* __restrict__ proj_b1,
           const float* __restrict__ proj_w2, const float* __restrict__ proj_b2,
           const float* __restrict__ ci_w1, const float* __restrict__ ci_b1,
           const float* __restrict__ ci_w2, const float* __restrict__ ci_b2,
           float* __restrict__ cscore, float* __restrict__ cgate) {
    __shared__ float P[64], A[64], h1[32], h2[16];
    const int n = blockIdx.x, t = threadIdx.x;
    P[t] = pooled[n * HID + t];
    A[t] = avg[n * HID + t];
    __syncthreads();
    if (t < 32) {
        float acc = proj_b1[t];
        for (int c = 0; c < HID; ++c) acc += P[c] * proj_w1[t * HID + c];
        h1[t] = gelu_f(acc);
    }
    if (t < 16) {
        float acc = ci_b1[t];
        for (int c = 0; c < HID; ++c) acc += A[c] * ci_w1[t * HID + c];
        h2[t] = gelu_f(acc);
    }
    __syncthreads();
    float acc1 = proj_b2[t];
    for (int j = 0; j < 32; ++j) acc1 += h1[j] * proj_w2[t * 32 + j];
    cscore[n * HID + t] = acc1;
    float acc2 = ci_b2[t];
    for (int j = 0; j < 16; ++j) acc2 += h2[j] * ci_w2[t * 16 + j];
    cgate[n * HID + t] = sigmoid_f(acc2);
}

// ---------------------------------------------------------------------------
// K7a: T[m][c] = cs * (cg*local + sg*global)   (row-major operand for tail GEMM)
// ---------------------------------------------------------------------------
__global__ void __launch_bounds__(256)
mix_kernel(const float* __restrict__ local_x, const float* __restrict__ global_x,
           const float* __restrict__ sg, const float* __restrict__ cgate,
           const float* __restrict__ cscore, float* __restrict__ T) {
    const int idx = blockIdx.x * 256 + threadIdx.x;   // 32768 pixels
    const int n = idx >> 12, hw = idx & 4095;
    const float s = sg[idx];
    const float* lp = local_x  + ((size_t)n << 18) + hw;
    const float* gp = global_x + ((size_t)n << 18) + hw;
    float* tp = T + (size_t)idx * HID;
    for (int c = 0; c < HID; ++c) {
        const float sx = cgate[n * HID + c] * lp[(size_t)c << 12]
                       + s * gp[(size_t)c << 12];
        tp[c] = cscore[n * HID + c] * sx;
    }
}

// ---------------------------------------------------------------------------
// K7b: tail GEMM  out[32768,896] = T[32768,64] @ tail_w^T  (NHWC output)
//      One wave owns one m-tile: A fragments (full K=64) loaded once, then
//      all 56 d-tiles are swept (T read exactly once; tail_w from L2).
// ---------------------------------------------------------------------------
__global__ void __launch_bounds__(256)
tail_gemm_kernel(const float* __restrict__ T, const float* __restrict__ Wt,
                 float* __restrict__ out) {
    const int lane = threadIdx.x & 31;
    const int wave = threadIdx.x >> 5;
    const int mt = blockIdx.x * 8 + wave;     // 2048 m-tiles
    const int m0 = mt << 4;

    const float* abase = T + (size_t)m0 * HID;
    const v16h a0 = load_frag_f32(abase,      HID, lane);
    const v16h a1 = load_frag_f32(abase + 32, HID, lane);

    const int mrow = m0 + ((lane < 16) ? 0 : 8);
    float* dstbase = out + (size_t)mrow * DMODEL + (lane & 15);

    for (int nt = 0; nt < 56; ++nt) {
        const float* bb = Wt + ((size_t)nt << 4) * HID;   // tail_w [896][64]
        v16h b0 = load_frag_f32(bb,      HID, lane);
        v16h b1 = load_frag_f32(bb + 32, HID, lane);
        v8f acc = {0.f,0.f,0.f,0.f,0.f,0.f,0.f,0.f};
        acc = __builtin_amdgcn_wmma_f32_16x16x32_f16(false, a0, false, b0,
                                                     (short)0, acc, false, false);
        acc = __builtin_amdgcn_wmma_f32_16x16x32_f16(false, a1, false, b1,
                                                     (short)0, acc, false, false);
        float* dst = dstbase + (nt << 4);
#pragma unroll
        for (int r = 0; r < 8; ++r) dst[(size_t)r * DMODEL] = acc[r];
    }
}

// ---------------------------------------------------------------------------
// Launch
// ---------------------------------------------------------------------------
extern "C" void kernel_launch(void* const* d_in, const int* in_sizes, int n_in,
                              void* d_out, int out_size, void* d_ws, size_t ws_size,
                              hipStream_t stream) {
    const float* x          = (const float*)d_in[0];
    const float* head_w     = (const float*)d_in[1];
    const float* head_b     = (const float*)d_in[2];
    const float* bn_w       = (const float*)d_in[3];
    const float* bn_b       = (const float*)d_in[4];
    const float* bn_mean    = (const float*)d_in[5];
    const float* bn_var     = (const float*)d_in[6];
    const float* amp_w      = (const float*)d_in[7];
    const float* amp_b      = (const float*)d_in[8];
    const float* pha_w      = (const float*)d_in[9];
    const float* pha_b      = (const float*)d_in[10];
    const float* conv_A     = (const float*)d_in[11];
    const float* conv_B     = (const float*)d_in[12];
    const float* conv_bias  = (const float*)d_in[13];
    const float* compress_w = (const float*)d_in[14];
    const float* compress_b = (const float*)d_in[15];
    const float* proj_w1    = (const float*)d_in[16];
    const float* proj_b1    = (const float*)d_in[17];
    const float* proj_w2    = (const float*)d_in[18];
    const float* proj_b2    = (const float*)d_in[19];
    const float* ci_w1      = (const float*)d_in[20];
    const float* ci_b1      = (const float*)d_in[21];
    const float* ci_w2      = (const float*)d_in[22];
    const float* ci_b2      = (const float*)d_in[23];
    const float* si_w       = (const float*)d_in[24];
    const float* si_b       = (const float*)d_in[25];
    const float* tail_w     = (const float*)d_in[26];
    float* out = (float*)d_out;

    float* ws = (float*)d_ws;
    float* xc   = ws;                 // 2,097,152 f   (reused later as T)
    float* gx   = ws + 2097152;       // 2,097,152 f
    float* lx   = ws + 4194304;       // 2,097,152 f
    float* sg   = ws + 6291456;       // 32768 f
    float* avg  = ws + 6324224;       // 512 f
    float* pool = ws + 6324736;       // 512 f
    float* cs   = ws + 6325248;       // 512 f
    float* cg   = ws + 6325760;       // 512 f
    float* w9   = ws + 6326272;       // 576 f
    float* bns  = ws + 6326848;       // 64 f
    float* twc  = ws + 6326912;       // 64 f
    float* tws  = ws + 6326976;       // 64 f
    float* Tm   = xc;                 // reuse xc buffer for mixed operand

    prep_kernel<<<1, 64, 0, stream>>>(bn_w, bn_var, conv_A, conv_B,
                                      bns, w9, twc, tws);

    head_gemm_kernel<<<256, 256, 0, stream>>>(x, head_w, head_b,
                                              bns, bn_mean, bn_b, xc);

    fft_branch_kernel<<<512, 256, 0, stream>>>(xc, amp_w, amp_b, pha_w, pha_b,
                                               twc, tws, gx, avg);

    dwconv_kernel<<<8192, 256, 0, stream>>>(xc, w9, conv_bias, lx);

    pool_kernel<<<NBATCH, 256, 0, stream>>>(xc, compress_w, compress_b, pool);

    sgate_kernel<<<128, 256, 0, stream>>>(lx, si_w, si_b, sg);

    mlp_kernel<<<NBATCH, 64, 0, stream>>>(pool, avg,
                                          proj_w1, proj_b1, proj_w2, proj_b2,
                                          ci_w1, ci_b1, ci_w2, ci_b2,
                                          cs, cg);

    mix_kernel<<<128, 256, 0, stream>>>(lx, gx, sg, cg, cs, Tm);

    tail_gemm_kernel<<<256, 256, 0, stream>>>(Tm, tail_w, out);
}
